// SSE_44126493999141
// MI455X (gfx1250) — matrix operand order repaired and checked by reading, not compiled, not run on hardware
//
#include <hip/hip_runtime.h>

typedef __attribute__((ext_vector_type(16))) _Float16 v16h;
typedef __attribute__((ext_vector_type(8)))  _Float16 v8h;
typedef __attribute__((ext_vector_type(8)))  float    v8f;
typedef unsigned int u32x4 __attribute__((ext_vector_type(4)));
typedef int          i32x8 __attribute__((ext_vector_type(8)));
typedef int          i32x4 __attribute__((ext_vector_type(4)));

#define B_  64
#define T_  256
#define D_  1024
#define M_  (B_ * T_)   // 16384 rows

#define BM 128
#define BN 64
#define BK 32
#define BPITCH 40       // padded LDS pitch (halfs) for transposed B tiles

// ---- TDM availability / arity probe (compile-time) ----
#if defined(__has_builtin)
#if __has_builtin(__builtin_amdgcn_tensor_load_to_lds) && \
    __has_builtin(__builtin_amdgcn_s_wait_tensorcnt)
#define HAS_TDM 1
#endif
#endif
#ifndef HAS_TDM
#define HAS_TDM 0
#endif
#if __has_include(<hip/amd_detail/amd_gfx1250_TDM.h>)
#define TDM_6ARG 1
#else
#define TDM_6ARG 0
#endif

#if HAS_TDM
// Issue a TDM 2D tile load (global -> LDS), f16 elements.
// D# per CDNA5 ISA 8.3/8.4: count=1, data_size=1 (2B), type=2 ("image").
__device__ __forceinline__ void tdm_load_2d_f16(const _Float16* gsrc, void* ldsdst,
                                                unsigned tile0, unsigned tile1,
                                                unsigned tensor_dim0, unsigned tensor_dim1,
                                                unsigned stride0)
{
    const unsigned long long ga = (unsigned long long)(uintptr_t)gsrc;
    const unsigned lds = (unsigned)(uintptr_t)ldsdst;   // addr[31:0] = LDS offset
    u32x4 g0;
    g0[0] = 1u;                                          // count=1 (user D#)
    g0[1] = lds;                                         // lds_addr
    g0[2] = (unsigned)ga;                                // global_addr[31:0]
    g0[3] = ((unsigned)(ga >> 32) & 0x1FFFFFFu) | (2u << 30);  // addr[56:32] | type=2
    i32x8 g1;
    g1[0] = (int)(1u << 16);                             // data_size=1 (2 bytes)
    g1[1] = (int)((tensor_dim0 & 0xFFFFu) << 16);
    g1[2] = (int)((tensor_dim0 >> 16) | ((tensor_dim1 & 0xFFFFu) << 16));
    g1[3] = (int)((tensor_dim1 >> 16) | (tile0 << 16));  // tile_dim0
    g1[4] = (int)(tile1 & 0xFFFFu);                      // tile_dim1 (tile_dim2=0)
    g1[5] = (int)stride0;                                // dim0 stride lo32
    g1[6] = 0;
    g1[7] = 0;
    const i32x4 z4 = {0, 0, 0, 0};
#if TDM_6ARG
    const i32x8 z8 = {0, 0, 0, 0, 0, 0, 0, 0};
    __builtin_amdgcn_tensor_load_to_lds(g0, g1, z4, z4, z8, 0);
#else
    __builtin_amdgcn_tensor_load_to_lds(g0, g1, z4, z4, 0);
#endif
}
#endif

__device__ inline v16h cat8(v8h a, v8h b) {
    return __builtin_shufflevector(a, b, 0,1,2,3,4,5,6,7,8,9,10,11,12,13,14,15);
}

struct __align__(8) H4 { _Float16 h[4]; };

// x -> hi (f16) + lo (f16) elementwise, float4-vectorized
__global__ __launch_bounds__(256)
void split_f16(const float* __restrict__ x, _Float16* __restrict__ hi,
               _Float16* __restrict__ lo, int n4)
{
    const int i = blockIdx.x * 256 + threadIdx.x;
    if (i >= n4) return;
    const float4 v = ((const float4*)x)[i];
    H4 hh, ll;
    hh.h[0] = (_Float16)v.x; ll.h[0] = (_Float16)(v.x - (float)hh.h[0]);
    hh.h[1] = (_Float16)v.y; ll.h[1] = (_Float16)(v.y - (float)hh.h[1]);
    hh.h[2] = (_Float16)v.z; ll.h[2] = (_Float16)(v.z - (float)hh.h[2]);
    hh.h[3] = (_Float16)v.w; ll.h[3] = (_Float16)(v.w - (float)hh.h[3]);
    *(H4*)(hi + 4 * (size_t)i) = hh;
    *(H4*)(lo + 4 * (size_t)i) = ll;
}

// fp32 GEMM via f16-split WMMA with double-buffered (ping-pong) staging.
// A and B both pre-split hi/lo f16. A tiles staged by the Tensor Data Mover.
// A@B ~= Ahi*Bhi + Alo*Bhi + Ahi*Blo (f32 acc).
// MODE 0: out = relu(A@W + bias); also writes hi/lo split of out
// MODE 1: out = (A@W + bias) * scale[n]
// MODE 2: out = relu((A@W + bias) * vin) + fin
template <int MODE>
__global__ __launch_bounds__(256)
void gemm_wmma(const _Float16* __restrict__ Ah, const _Float16* __restrict__ Al,
               const _Float16* __restrict__ Bh, const _Float16* __restrict__ Bl,
               const float* __restrict__ bias, const float* __restrict__ scale,
               const float* __restrict__ vin, const float* __restrict__ fin,
               float* __restrict__ out,
               _Float16* __restrict__ outh, _Float16* __restrict__ outl)
{
    // double-buffered staging: A = 2 x (hi 8KB + lo 8KB); B = 2 x transposed hi/lo
    __shared__ __align__(16) unsigned char smemA[2][BM * BK * 4];   // 32 KB
    __shared__ __align__(16) _Float16 sBth[2][BN * BPITCH];         // 10 KB
    __shared__ __align__(16) _Float16 sBtl[2][BN * BPITCH];         // 10 KB

    const int tid  = threadIdx.x;
    const int wv   = tid >> 5;
    const int lane = tid & 31;
    const int m0   = blockIdx.x * BM;
    const int n0   = blockIdx.y * BN;

    v8f acc[4] = {};

    // per-thread staging registers (global -> reg -> LDS)
    float2 rBh[2], rBl[2];
#if !HAS_TDM
    float4 rAh2[2], rAl2[2];    // fallback if no TDM builtin
#endif

    auto gloadB = [&](int k0) {
        #pragma unroll
        for (int i = 0; i < 2; ++i) {
            const int q  = tid + i * 256;
            const int r  = q >> 4;
            const int c4 = (q & 15) << 2;
            rBh[i] = *(const float2*)(Bh + (size_t)(k0 + r) * D_ + n0 + c4);
            rBl[i] = *(const float2*)(Bl + (size_t)(k0 + r) * D_ + n0 + c4);
        }
    };
    auto storeB = [&](int buf) {
        #pragma unroll
        for (int i = 0; i < 2; ++i) {
            const int q  = tid + i * 256;
            const int r  = q >> 4;
            const int c4 = (q & 15) << 2;
            union { float2 f2; _Float16 h[4]; } uh, ul;
            uh.f2 = rBh[i]; ul.f2 = rBl[i];
            #pragma unroll
            for (int e = 0; e < 4; ++e) {
                sBth[buf][(c4 + e) * BPITCH + r] = uh.h[e];
                sBtl[buf][(c4 + e) * BPITCH + r] = ul.h[e];
            }
        }
    };
    auto stageA_issue = [&](int buf, int k0) {
#if HAS_TDM
        if (wv == 0) {   // one TDM issue per plane, wave-level DMA
            tdm_load_2d_f16(Ah + (size_t)m0 * D_ + k0,
                            (void*)smemA[buf], BK, BM, D_, M_, D_);
            tdm_load_2d_f16(Al + (size_t)m0 * D_ + k0,
                            (void*)(smemA[buf] + BM * BK * 2), BK, BM, D_, M_, D_);
        }
#else
        #pragma unroll
        for (int i = 0; i < 2; ++i) {
            const int q = tid + i * 256;
            const int r = q >> 2;
            const int c = (q & 3) << 3;
            rAh2[i] = *(const float4*)(Ah + (size_t)(m0 + r) * D_ + k0 + c);
            rAl2[i] = *(const float4*)(Al + (size_t)(m0 + r) * D_ + k0 + c);
        }
        (void)buf;
#endif
    };
    auto stageA_store = [&](int buf) {
#if !HAS_TDM
        _Float16* dAh = (_Float16*)smemA[buf];
        _Float16* dAl = (_Float16*)(smemA[buf] + BM * BK * 2);
        #pragma unroll
        for (int i = 0; i < 2; ++i) {
            const int q = tid + i * 256;
            const int r = q >> 2;
            const int c = (q & 3) << 3;
            *(float4*)&dAh[r * BK + c] = rAh2[i];
            *(float4*)&dAl[r * BK + c] = rAl2[i];
        }
#else
        (void)buf;
#endif
    };

    // ---- prologue: stage buffer 0 / chunk 0 ----
    stageA_issue(0, 0);
    gloadB(0);
    stageA_store(0);
    storeB(0);

    for (int k0 = 0; k0 < D_; k0 += BK) {
        const int  cur     = (k0 >> 5) & 1;
        const int  nxt     = cur ^ 1;
        const bool hasnext = (k0 + BK) < D_;

        // (1) kick off next chunk: TDM issue (A) + global loads to regs (B)
        if (hasnext) {
            stageA_issue(nxt, k0 + BK);
            gloadB(k0 + BK);
        }
#if HAS_TDM
        // (2) current chunk's 2 TDMs are the oldest in the in-order stream
        if (wv == 0) {
            if (hasnext) __builtin_amdgcn_s_wait_tensorcnt(2);
            else         __builtin_amdgcn_s_wait_tensorcnt(0);
        }
#endif
        __syncthreads();   // (3) cur buffers now valid for everyone

        // ---- (4) compute from cur ----
        // A fragment: two contiguous 8-half runs per lane
        // k(h) = (h&7) + ((h&8)<<1) + ((lane&16)>>1)
        const int mrow  = wv * 16 + (lane & 15);
        const int kbase = (lane & 16) >> 1;
        const _Float16* sAh = (const _Float16*)smemA[cur];
        const _Float16* sAl = (const _Float16*)(smemA[cur] + BM * BK * 2);
        const _Float16* ap = sAh + mrow * BK + kbase;
        const v16h ahi = cat8(*(const v8h*)ap, *(const v8h*)(ap + 16));
        const _Float16* alp = sAl + mrow * BK + kbase;
        const v16h alo = cat8(*(const v8h*)alp, *(const v8h*)(alp + 16));

        // B fragments: two contiguous 8-half runs per lane; k(h) = (h&7) + ((h&8)<<1) + ((lane&16)>>1)
        const int kb = (lane & 16) >> 1;
        #pragma unroll
        for (int nf = 0; nf < 4; ++nf) {
            const int ncol = nf * 16 + (lane & 15);
            const _Float16* bp = &sBth[cur][ncol * BPITCH + kb];
            const v16h bhi = cat8(*(const v8h*)bp, *(const v8h*)(bp + 16));
            const _Float16* bq = &sBtl[cur][ncol * BPITCH + kb];
            const v16h blo = cat8(*(const v8h*)bq, *(const v8h*)(bq + 16));
            acc[nf] = __builtin_amdgcn_wmma_f32_16x16x32_f16(
                false, ahi, false, bhi, (short)0, acc[nf], false, false);
            acc[nf] = __builtin_amdgcn_wmma_f32_16x16x32_f16(
                false, alo, false, bhi, (short)0, acc[nf], false, false);
            acc[nf] = __builtin_amdgcn_wmma_f32_16x16x32_f16(
                false, ahi, false, blo, (short)0, acc[nf], false, false);
        }

        // drain next chunk's register staging into LDS while WMMAs retire
        if (hasnext) {
            stageA_store(nxt);
            storeB(nxt);
        }
        __syncthreads();   // (5) nxt staged; cur free for overwrite next iter
    }

    // Epilogue. C/D layout: VGPR r -> m = r + 8*(lane>=16), n = lane&15.
    #pragma unroll
    for (int nf = 0; nf < 4; ++nf) {
        const int ng = n0 + nf * 16 + (lane & 15);
        #pragma unroll
        for (int r = 0; r < 8; ++r) {
            const int mg = m0 + wv * 16 + r + ((lane & 16) ? 8 : 0);
            const size_t idx = (size_t)mg * D_ + ng;
            float val = acc[nf][r] + bias[ng];
            if (MODE == 0) {
                val = fmaxf(val, 0.f);
                out[idx] = val;
                const _Float16 h = (_Float16)val;
                outh[idx] = h;
                outl[idx] = (_Float16)(val - (float)h);
            } else if (MODE == 1) {
                out[idx] = val * scale[ng];
            } else {
                out[idx] = fmaxf(val * vin[idx], 0.f) + fin[idx];
            }
        }
    }
}

// Sequential per-sample scan. One workgroup (256 threads, 8 waves) per sample.
__global__ __launch_bounds__(256)
void sse_scan(const float* __restrict__ f, const float* __restrict__ u,
              const int* __restrict__ mask, const int* __restrict__ um,
              const float* __restrict__ bWp, float* __restrict__ v)
{
    const int b    = blockIdx.x;
    const int tid  = threadIdx.x;
    const int wv   = tid >> 5;
    const int lane = tid & 31;

    const float* cf  = f    + (size_t)b * T_ * D_;
    const float* cu  = u    + (size_t)b * T_ * D_;
    const int*   cm  = mask + (size_t)b * T_ * T_;
    const int*   cum = um   + (size_t)b * T_;
    float*       cv  = v    + (size_t)b * T_ * D_;

    __shared__ float uj[D_];
    __shared__ float sc[T_];
    __shared__ float red[256];
    __shared__ float salpha[T_];
    __shared__ int   ired[256];
    __shared__ int   s_stop;

    float4* cv4 = (float4*)cv;
    for (int i = tid; i < (T_ * D_) / 4; i += 256)
        cv4[i] = make_float4(0.f, 0.f, 0.f, 0.f);

    for (int d = tid; d < D_; d += 256) cv[d] = cf[d];

    const int m00 = cm[0];
    ired[tid] = (tid > 0 && cm[tid] != m00) ? tid : T_;
    if (tid == 0) s_stop = 0;
    __syncthreads();
    for (int s = 128; s > 0; s >>= 1) {
        if (tid < s) ired[tid] = min(ired[tid], ired[tid + s]);
        __syncthreads();
    }
    const int kmin = ired[0];
    if (kmin < T_) {
        for (int d = tid; d < D_; d += 256)
            cv[(size_t)kmin * D_ + d] = cf[(size_t)kmin * D_ + d];
    }
    __threadfence_block();
    __syncthreads();

    const float bW = bWp[0];

    for (int j = 1; j < T_; ++j) {
        if (tid == 0 && cum[j] == 0) s_stop = 1;
        for (int d = tid; d < D_; d += 256) uj[d] = cu[(size_t)j * D_ + d];
        ired[tid] = (tid < j && cm[(size_t)j * T_ + tid] == 1) ? tid : -1;
        __syncthreads();
        for (int s = 128; s > 0; s >>= 1) {
            if (tid < s) ired[tid] = max(ired[tid], ired[tid + s]);
            __syncthreads();
        }
        const int l       = ired[0];
        const int stopped = s_stop;
        __syncthreads();
        if (stopped) continue;

        if (l < 0) {
            for (int d = tid; d < D_; d += 256)
                cv[(size_t)j * D_ + d] = cf[(size_t)j * D_ + d];
            __threadfence_block();
            __syncthreads();
            continue;
        }

        const int L = j - l;
        for (int i = l + wv; i < j; i += 8) {
            float s = 0.f;
            for (int d = lane; d < D_; d += 32)
                s += cv[(size_t)i * D_ + d] * uj[d];
            #pragma unroll
            for (int off = 16; off > 0; off >>= 1)
                s += __shfl_down(s, off, 32);
            if (lane == 0) sc[i - l] = s + bW;
        }
        __syncthreads();

        const float val = (tid < L) ? sc[tid] : -1e30f;
        red[tid] = val;
        __syncthreads();
        for (int s = 128; s > 0; s >>= 1) {
            if (tid < s) red[tid] = fmaxf(red[tid], red[tid + s]);
            __syncthreads();
        }
        const float mx = red[0];
        __syncthreads();
        const float e = (tid < L) ? __expf(val - mx) : 0.f;
        red[tid] = e;
        __syncthreads();
        for (int s = 128; s > 0; s >>= 1) {
            if (tid < s) red[tid] += red[tid + s];
            __syncthreads();
        }
        const float ssum = red[0];
        __syncthreads();
        salpha[tid] = e / ssum;
        __syncthreads();

        for (int d = tid; d < D_; d += 256) {
            float a = 0.f;
            for (int i = 0; i < L; ++i)
                a += salpha[i] * cv[(size_t)(l + i) * D_ + d];
            cv[(size_t)j * D_ + d] = tanhf(a);
        }
        __threadfence_block();
        __syncthreads();
    }
}

extern "C" void kernel_launch(void* const* d_in, const int* in_sizes, int n_in,
                              void* d_out, int out_size, void* d_ws, size_t ws_size,
                              hipStream_t stream) {
    (void)in_sizes; (void)n_in; (void)out_size; (void)ws_size;

    const float* feature = (const float*)d_in[0];
    const int*   mask    = (const int*)  d_in[1];
    const int*   umask   = (const int*)  d_in[2];
    const float* Wi      = (const float*)d_in[3];
    const float* bi      = (const float*)d_in[4];
    const float* Wq      = (const float*)d_in[5];
    const float* bq      = (const float*)d_in[6];
    const float* wW      = (const float*)d_in[7];
    const float* bW      = (const float*)d_in[8];
    const float* Wo      = (const float*)d_in[9];
    const float* bo      = (const float*)d_in[10];
    float* out = (float*)d_out;

    const size_t MB = 1024u * 1024u;
    char* ws = (char*)d_ws;
    // layout (204 MB): f | u (transient: feature hi/lo) | fh | fl | weight splits
    float*    f_ws  = (float*)   (ws + 0);         // 64 MB fp32 f
    float*    u_ws  = (float*)   (ws + 64 * MB);   // 64 MB fp32 u (after GEMM1)
    _Float16* featH = (_Float16*)(ws + 64 * MB);   // 32 MB (dead after GEMM1)
    _Float16* featL = (_Float16*)(ws + 96 * MB);   // 32 MB (dead after GEMM1)
    _Float16* fh    = (_Float16*)(ws + 128 * MB);  // 32 MB f hi
    _Float16* fl    = (_Float16*)(ws + 160 * MB);  // 32 MB f lo
    _Float16* Wih   = (_Float16*)(ws + 192 * MB);
    _Float16* Wil   = (_Float16*)(ws + 194 * MB);
    _Float16* Wqh   = (_Float16*)(ws + 196 * MB);
    _Float16* Wql   = (_Float16*)(ws + 198 * MB);
    _Float16* Woh   = (_Float16*)(ws + 200 * MB);
    _Float16* Wol   = (_Float16*)(ws + 202 * MB);

    // pre-split feature and weights into hi/lo f16 planes
    const int n4f = (M_ * D_) / 4;
    split_f16<<<(n4f + 255) / 256, 256, 0, stream>>>(feature, featH, featL, n4f);
    const int n4w = (D_ * D_) / 4;
    const int gsw = (n4w + 255) / 256;
    split_f16<<<gsw, 256, 0, stream>>>(Wi, Wih, Wil, n4w);
    split_f16<<<gsw, 256, 0, stream>>>(Wq, Wqh, Wql, n4w);
    split_f16<<<gsw, 256, 0, stream>>>(Wo, Woh, Wol, n4w);

    dim3 ggrid(M_ / BM, D_ / BN);   // 128 x 16 blocks
    gemm_wmma<0><<<ggrid, 256, 0, stream>>>(
        featH, featL, Wih, Wil, bi, nullptr, nullptr, nullptr,
        f_ws, fh, fl);
    gemm_wmma<1><<<ggrid, 256, 0, stream>>>(
        fh, fl, Wqh, Wql, bq, wW, nullptr, nullptr,
        u_ws, nullptr, nullptr);
    sse_scan<<<B_, 256, 0, stream>>>(f_ws, u_ws, mask, umask, bW, out);
    gemm_wmma<2><<<ggrid, 256, 0, stream>>>(
        fh, fl, Woh, Wol, bo, nullptr, out, f_ws,
        out, nullptr, nullptr);
}
